// QLoRALinear_4672924418482
// MI455X (gfx1250) — matrix operand order repaired and checked by reading, not compiled, not run on hardware
//
#include <hip/hip_runtime.h>

// ---------------------------------------------------------------------------
// QLoRA linear for MI455X (gfx1250, wave32, WMMA):
//   out = x @ W_eff^T + bias,   W_eff = NF4dequant(codes)*absmax + s*(A@B)^T
// Pass 1: build W_eff as bf16 (L2-resident, 33.5 MB)
// Pass 2: convert x to bf16
// Pass 3: tiled bf16 WMMA GEMM with f32 accumulation + bias epilogue
// ---------------------------------------------------------------------------

typedef __attribute__((ext_vector_type(8)))  __bf16 v8bf;
typedef __attribute__((ext_vector_type(16))) __bf16 v16bf;
typedef __attribute__((ext_vector_type(8)))  float  v8f;

#define TOK   8192
#define DIN   4096
#define DOUT  4096
#define NFBLK 64
#define SCALING 4.0f   // LORA_ALPHA / sqrt(RANK) = 16/4

#define BM 128
#define BN 128
#define BK 64
#define LDSTR 72   // padded LDS row stride in bf16 (144 B, 16B-aligned, conflict-avoiding)

__device__ __constant__ float NF4_CODE[16] = {
    -1.0f, -0.6961928009986877f, -0.5250730514526367f, -0.39491748809814453f,
    -0.28444138169288635f, -0.18477343022823334f, -0.09105003625154495f, 0.0f,
    0.07958029955625534f, 0.16093020141124725f, 0.24611230194568634f,
    0.33791524171829224f, 0.44070982933044434f, 0.5626170039176941f,
    0.7229568362236023f, 1.0f};

__device__ __forceinline__ v16bf cat16(v8bf lo, v8bf hi) {
  return __builtin_shufflevector(lo, hi, 0,1,2,3,4,5,6,7,8,9,10,11,12,13,14,15);
}

// ---------------------------------------------------------------------------
// Pass 1: W_eff[o, i] = NF4[codes[o,i]] * absmax[o, i/64]
//                     + SCALING * dot(lora_A[i, 0:16], lora_B[0:16, o])
// One block per output row o; 256 threads x 16 elements = 4096.
// ---------------------------------------------------------------------------
__global__ __launch_bounds__(256)
void build_weff_kernel(const int*   __restrict__ codes,
                       const float* __restrict__ absmax,
                       const float* __restrict__ lora_A,
                       const float* __restrict__ lora_B,
                       __bf16*      __restrict__ weff) {
  __shared__ float lut[16];
  __shared__ float bcol[16];
  const int o = blockIdx.x;
  const int t = threadIdx.x;
  if (t < 16) {
    lut[t]  = NF4_CODE[t];
    bcol[t] = lora_B[(size_t)t * DOUT + o];
  }
  __syncthreads();

  const int i0 = t * 16;                                // 16 elems / thread
  const float am = absmax[(size_t)o * (DIN / NFBLK) + (i0 >> 6)];
  const int4* cp = (const int4*)(codes + (size_t)o * DIN + i0);

  alignas(16) __bf16 outv[16];
  #pragma unroll
  for (int c = 0; c < 4; ++c) {
    int4 cc = cp[c];
    int ids[4] = {cc.x, cc.y, cc.z, cc.w};
    #pragma unroll
    for (int j = 0; j < 4; ++j) {
      const int i = i0 + c * 4 + j;
      const float w = lut[ids[j] & 15] * am;
      const float4* ar = (const float4*)(lora_A + (size_t)i * 16);
      float dot = 0.f;
      #pragma unroll
      for (int q = 0; q < 4; ++q) {
        float4 a4 = ar[q];
        dot += a4.x * bcol[q*4+0] + a4.y * bcol[q*4+1]
             + a4.z * bcol[q*4+2] + a4.w * bcol[q*4+3];
      }
      outv[c * 4 + j] = (__bf16)(w + SCALING * dot);
    }
  }
  uint4* dst = (uint4*)(weff + (size_t)o * DIN + i0);
  dst[0] = *(const uint4*)(&outv[0]);
  dst[1] = *(const uint4*)(&outv[8]);
}

// ---------------------------------------------------------------------------
// Pass 2: fp32 -> bf16 conversion of x (8 elements per thread)
// ---------------------------------------------------------------------------
__global__ __launch_bounds__(256)
void convert_x_kernel(const float* __restrict__ x, __bf16* __restrict__ xb) {
  const size_t idx = ((size_t)blockIdx.x * blockDim.x + threadIdx.x) * 8;
  float4 a = *(const float4*)(x + idx);
  float4 b = *(const float4*)(x + idx + 4);
  alignas(16) __bf16 o[8] = {(__bf16)a.x, (__bf16)a.y, (__bf16)a.z, (__bf16)a.w,
                             (__bf16)b.x, (__bf16)b.y, (__bf16)b.z, (__bf16)b.w};
  *(uint4*)(xb + idx) = *(const uint4*)o;
}

// ---------------------------------------------------------------------------
// Pass 3: C[TOK, DOUT] = Xb[TOK, DIN] * Weff[DOUT, DIN]^T + bias
// Block: 256 threads = 8 waves; block tile 128x128; wave tile 64x32
// (waves laid out 2(M) x 4(N)); 16x16x32 bf16 WMMA, f32 accum.
// ---------------------------------------------------------------------------
__global__ __launch_bounds__(256)
void qlora_gemm_kernel(const __bf16* __restrict__ xb,
                       const __bf16* __restrict__ weff,
                       const float*  __restrict__ bias,
                       float*        __restrict__ out) {
  __shared__ alignas(16) __bf16 ldsA[BM][LDSTR];
  __shared__ alignas(16) __bf16 ldsB[BN][LDSTR];

  const int t    = threadIdx.x;
  const int lane = t & 31;
  const int w    = t >> 5;      // 0..7
  const int wm   = w & 1;       // M wave offset: wm*64
  const int wn   = w >> 1;      // N wave offset: wn*32
  const int half = lane >> 4;   // 0/1
  const int ln   = lane & 15;

  const int bN = blockIdx.x;
  const int bM = blockIdx.y;

  // cooperative tile load mapping: 2 threads per row, 32 bf16 (64B) each
  const int lrow = t >> 1;              // 0..127
  const int lkc  = (t & 1) * 32;        // 0 or 32

  const __bf16* gA = xb   + (size_t)(bM * BM + lrow) * DIN + lkc;
  const __bf16* gB = weff + (size_t)(bN * BN + lrow) * DIN + lkc;

  v8f acc[4][2];
  #pragma unroll
  for (int m = 0; m < 4; ++m)
    #pragma unroll
    for (int n = 0; n < 2; ++n) { v8f z = {}; acc[m][n] = z; }

  for (int k0 = 0; k0 < DIN; k0 += BK) {
    if (k0 + 2 * BK <= DIN - BK) {
      __builtin_prefetch(gA + k0 + 2 * BK, 0, 0);
      __builtin_prefetch(gB + k0 + 2 * BK, 0, 0);
    }
    // global -> LDS stage (64 B per thread per operand)
    {
      const uint4* pa = (const uint4*)(gA + k0);
      const uint4* pb = (const uint4*)(gB + k0);
      uint4 ra0 = pa[0], ra1 = pa[1], ra2 = pa[2], ra3 = pa[3];
      uint4 rb0 = pb[0], rb1 = pb[1], rb2 = pb[2], rb3 = pb[3];
      uint4* sa = (uint4*)(&ldsA[lrow][lkc]);
      uint4* sb = (uint4*)(&ldsB[lrow][lkc]);
      sa[0] = ra0; sa[1] = ra1; sa[2] = ra2; sa[3] = ra3;
      sb[0] = rb0; sb[1] = rb1; sb[2] = rb2; sb[3] = rb3;
    }
    __syncthreads();

    #pragma unroll
    for (int kk = 0; kk < 2; ++kk) {
      const int kb = kk * 32;
      // A fragments: lane l<16 -> row l, K {0..7,16..23}; l>=16 -> K {8..15,24..31}
      v16bf af[4];
      #pragma unroll
      for (int m = 0; m < 4; ++m) {
        const __bf16* p = &ldsA[wm * 64 + m * 16 + ln][kb + half * 8];
        af[m] = cat16(*(const v8bf*)p, *(const v8bf*)(p + 16));
      }
      // B fragments: lane n<16 -> col n, K {0..7,16..23}; n>=16 -> col n-16, K {8..15,24..31}
      v16bf bfr[2];
      #pragma unroll
      for (int n = 0; n < 2; ++n) {
        const __bf16* p = &ldsB[wn * 32 + n * 16 + ln][kb + half * 8];
        bfr[n] = cat16(*(const v8bf*)p, *(const v8bf*)(p + 16));
      }
      #pragma unroll
      for (int m = 0; m < 4; ++m)
        #pragma unroll
        for (int n = 0; n < 2; ++n)
          acc[m][n] = __builtin_amdgcn_wmma_f32_16x16x32_bf16(
              false, af[m], false, bfr[n], (short)0, acc[m][n], false, false);
    }
    __syncthreads();
  }

  // Epilogue: C layout -> VGPR r holds row (half*8 + r), col = ln
  #pragma unroll
  for (int m = 0; m < 4; ++m) {
    #pragma unroll
    for (int n = 0; n < 2; ++n) {
      const int col = bN * BN + wn * 32 + n * 16 + ln;
      const float bv = bias[col];
      #pragma unroll
      for (int r = 0; r < 8; ++r) {
        const int rowO = bM * BM + wm * 64 + m * 16 + half * 8 + r;
        out[(size_t)rowO * DOUT + col] = acc[m][n][r] + bv;
      }
    }
  }
}

// ---------------------------------------------------------------------------
// Host launcher
// ---------------------------------------------------------------------------
extern "C" void kernel_launch(void* const* d_in, const int* in_sizes, int n_in,
                              void* d_out, int out_size, void* d_ws, size_t ws_size,
                              hipStream_t stream) {
  const float* x       = (const float*)d_in[0];
  const int*   w_codes = (const int*)  d_in[1];
  const float* absmax  = (const float*)d_in[2];
  const float* bias    = (const float*)d_in[3];
  const float* lora_A  = (const float*)d_in[4];
  const float* lora_B  = (const float*)d_in[5];
  float*       out     = (float*)d_out;

  __bf16* weff = (__bf16*)d_ws;                                  // 33.5 MB
  __bf16* xb   = (__bf16*)((char*)d_ws + (size_t)DOUT * DIN * 2); // 67 MB

  build_weff_kernel<<<DOUT, 256, 0, stream>>>(w_codes, absmax, lora_A, lora_B, weff);

  const int conv_blocks = (TOK * DIN) / (256 * 8);
  convert_x_kernel<<<conv_blocks, 256, 0, stream>>>(x, xb);

  dim3 grid(DOUT / BN, TOK / BM);
  qlora_gemm_kernel<<<grid, 256, 0, stream>>>(xb, weff, bias, out);
}
